// OOTGSetConv_86251533238889
// MI455X (gfx1250) — matrix operand order repaired and checked by reading, not compiled, not run on hardware
//
#include <hip/hip_runtime.h>

typedef __attribute__((ext_vector_type(16))) _Float16 v16h;
typedef __attribute__((ext_vector_type(2)))  _Float16 v2h;
typedef __attribute__((ext_vector_type(8)))  float    v8f;

#define M_BATCH 2
#define N_PTS   4096
#define DX      2
#define DZ      64
#define G_PTS   16384          // 128*128 grid points per batch
#define ROWS_PER_BLOCK 128     // grid points (output rows) per workgroup
#define KC      128            // K (n-points) per staged chunk
#define KSTEP   32             // K per v_wmma_f32_16x16x32_f16
#define NCHUNK  (N_PTS / KC)   // 32
#define ZB_ELTS ((KC / KSTEP) * 4 * 32 * 16)   // 8192 halfs per buffer

// Stage one K-chunk into LDS buffers: scaled coords + expc*|xs|^2, and z in
// WMMA-B fragment layout (fragment (ks,nt): lane holds 16 contiguous halfs,
// K = j + 16*(lane>=16), N = lane&15).
__device__ __forceinline__
void stage_chunk(const float* __restrict__ xbase, const float* __restrict__ zbase,
                 int k0, int tid,
                 float2* __restrict__ xs2, float* __restrict__ ee,
                 _Float16* __restrict__ zb,
                 float inv_ls0, float inv_ls1, float expc)
{
    for (int i = tid; i < KC / 2; i += 256) {             // 64 float4s
        float4 v = *(const float4*)(xbase + (size_t)k0 * DX + i * 4);
        float s00 = v.x * inv_ls0, s01 = v.y * inv_ls1;
        float s10 = v.z * inv_ls0, s11 = v.w * inv_ls1;
        xs2[2 * i]     = make_float2(s00, s01);
        xs2[2 * i + 1] = make_float2(s10, s11);
        ee[2 * i]      = expc * fmaf(s00, s00, s01 * s01);
        ee[2 * i + 1]  = expc * fmaf(s10, s10, s11 * s11);
    }
    for (int i = tid; i < (KC * DZ) / 4; i += 256) {      // 2048 float4s
        int k  = i >> 4;            // row within chunk, 0..127
        int c4 = (i & 15) << 2;     // column group: 0,4,...,60
        float4 v = *(const float4*)(zbase + (size_t)(k0 + k) * DZ + c4);
        float vv[4] = {v.x, v.y, v.z, v.w};
        #pragma unroll
        for (int q = 0; q < 4; ++q) {
            int col   = c4 + q;
            int nt    = col >> 4;           // 16-wide N tile
            int ks    = k >> 5;             // K step within chunk
            int kl    = k & 31;             // K within step
            int blane = (col & 15) | ((kl >> 4) << 4);
            int j     = kl & 15;
            zb[((((ks << 2) | nt) * 32) + blane) * 16 + j] = (_Float16)vv[q];
        }
    }
}

__global__ __launch_bounds__(256)
void ootg_setconv_wmma(const float* __restrict__ x,        // [m, n, dx]
                       const float* __restrict__ z,        // [m, n, dz]
                       const float* __restrict__ x_grid,   // [m, H, W, dx]
                       const float* __restrict__ z_grid,   // [m, H, W, dz]
                       const float* __restrict__ ls_param, // [dx]
                       float* __restrict__ out)            // [m, H, W, dz]
{
    // double-buffered staging: 2 x (16 KB z-fragments + 1 KB coords + 0.5 KB ee)
    __shared__ __align__(32) _Float16 zb[2 * ZB_ELTS];
    __shared__ float2 xs2[2 * KC];
    __shared__ float  ee[2 * KC];

    const int tid  = threadIdx.x;
    const int lane = tid & 31;
    const int wave = tid >> 5;

    const int batch  = blockIdx.x / (G_PTS / ROWS_PER_BLOCK);
    const int rowblk = blockIdx.x % (G_PTS / ROWS_PER_BLOCK);
    const int row0   = rowblk * ROWS_PER_BLOCK;

    // lengthscale = 1e-5 + softplus(param); fast path: one v_exp + one v_log
    const float inv_ls0 = 1.0f / (1e-5f + __logf(1.0f + __expf(ls_param[0])));
    const float inv_ls1 = 1.0f / (1e-5f + __logf(1.0f + __expf(ls_param[1])));

    const float expc = -0.72134752044448170368f;  // -0.5*log2(e)

    // per-lane grid point (A layout: both lane halves serve row M = lane&15)
    const int myrow = row0 + wave * 16 + (lane & 15);
    const float xg0 = x_grid[((size_t)batch * G_PTS + myrow) * DX + 0] * inv_ls0;
    const float xg1 = x_grid[((size_t)batch * G_PTS + myrow) * DX + 1] * inv_ls1;
    // expanded quadratic form: arg = cr + ee[n] + a0*xs0[n] + a1*xs1[n]
    const float cr = expc * (xg0 * xg0 + xg1 * xg1);
    const float a0 = -2.0f * expc * xg0;
    const float a1 = -2.0f * expc * xg1;

    v8f zero = {};
    v8f acc[4];
    #pragma unroll
    for (int t = 0; t < 4; ++t) acc[t] = zero;

    const float* xbase = x + (size_t)batch * N_PTS * DX;
    const float* zbase = z + (size_t)batch * N_PTS * DZ;

    // prologue: stage chunk 0 into buffer 0
    stage_chunk(xbase, zbase, 0, tid, xs2, ee, zb, inv_ls0, inv_ls1, expc);

    for (int c = 0; c < NCHUNK; ++c) {
        __syncthreads();   // staging of buffer c&1 complete; buffer (c+1)&1 free

        // overlap: stage next chunk into the other buffer while we compute
        if (c + 1 < NCHUNK) {
            int b1 = (c + 1) & 1;
            stage_chunk(xbase, zbase, (c + 1) * KC, tid,
                        xs2 + b1 * KC, ee + b1 * KC, zb + b1 * ZB_ELTS,
                        inv_ls0, inv_ls1, expc);
        }
        // prefetch chunk c+2 (32 KB = 256 x 128B lines, one per thread)
        if (c + 2 < NCHUNK)
            __builtin_prefetch(zbase + (size_t)(c + 2) * KC * DZ + tid * 32, 0, 0);

        const int b = c & 1;
        const float2*   cxs = xs2 + b * KC;
        const float*    cee = ee  + b * KC;
        const _Float16* czb = zb  + b * ZB_ELTS;

        #pragma unroll
        for (int ks = 0; ks < KC / KSTEP; ++ks) {
            // A fragment: RBF weights for 16 rows x 32 n-points.
            // consecutive half-indices j,j+1 -> consecutive K: pair the work.
            union { v16h v; v2h h[8]; } af;
            #pragma unroll
            for (int jj = 0; jj < 8; ++jj) {
                int j0 = jj * 2;
                int K0 = ((lane >> 4) << 4) + j0;
                int nl = ks * KSTEP + K0;                 // even -> 16B aligned
                float4 sv = *(const float4*)&cxs[nl];     // two float2 coords (b128)
                float2 ev = *(const float2*)&cee[nl];     // two ee terms (b64)
                float arg0 = fmaf(a0, sv.x, fmaf(a1, sv.y, ev.x + cr));
                float arg1 = fmaf(a0, sv.z, fmaf(a1, sv.w, ev.y + cr));
                float w0 = exp2f(arg0);                   // v_exp_f32
                float w1 = exp2f(arg1);
                auto pk = __builtin_amdgcn_cvt_pkrtz(w0, w1);  // v_cvt_pk_rtz_f16_f32
                af.h[jj] = *reinterpret_cast<v2h*>(&pk);
            }
            // 4 WMMAs cover the 64 output channels for this wave's 16 rows
            #pragma unroll
            for (int nt = 0; nt < 4; ++nt) {
                v16h bfrag = *(const v16h*)&czb[((((ks << 2) | nt) * 32) + lane) * 16];
                acc[nt] = __builtin_amdgcn_wmma_f32_16x16x32_f16(
                    /*neg_a=*/false, af.v, /*neg_b=*/false, bfrag,
                    /*c_mod=*/(short)0, acc[nt],
                    /*reuse_a=*/false, /*reuse_b=*/false);
            }
        }
    }

    // out = z_grid + acc.  C/D layout: VGPR v, lanes 0-15: M=v; lanes 16-31: M=8+v; N=lane&15
    #pragma unroll
    for (int nt = 0; nt < 4; ++nt) {
        #pragma unroll
        for (int v = 0; v < 8; ++v) {
            int r = row0 + wave * 16 + ((lane >> 4) << 3) + v;
            int cc = nt * 16 + (lane & 15);
            size_t idx = ((size_t)batch * G_PTS + r) * (size_t)DZ + cc;
            out[idx] = z_grid[idx] + acc[nt][v];
        }
    }
}

extern "C" void kernel_launch(void* const* d_in, const int* in_sizes, int n_in,
                              void* d_out, int out_size, void* d_ws, size_t ws_size,
                              hipStream_t stream) {
    const float* x  = (const float*)d_in[0];   // [2, 4096, 2]
    const float* z  = (const float*)d_in[1];   // [2, 4096, 64]
    const float* xg = (const float*)d_in[2];   // [2, 128, 128, 2]
    const float* zg = (const float*)d_in[3];   // [2, 128, 128, 64]
    const float* lp = (const float*)d_in[4];   // [2]
    float* out = (float*)d_out;                // [2, 128, 128, 64]

    dim3 grid(M_BATCH * (G_PTS / ROWS_PER_BLOCK));  // 256 blocks
    dim3 block(256);                                // 8 waves
    hipLaunchKernelGGL(ootg_setconv_wmma, grid, block, 0, stream,
                       x, z, xg, zg, lp, out);
}
